// diffusion_model_76802605187805
// MI455X (gfx1250) — compile-verified, matrix operand order checked by reading, not run on hardware
//
#include <hip/hip_runtime.h>

// ---------------------------------------------------------------------------
// Sizes (match reference)
// ---------------------------------------------------------------------------
#define DIM      64
#define N_TRACK  100000
#define N_INTER  50000
#define N_EDGE   800000

typedef __attribute__((ext_vector_type(16))) _Float16 v16h;
typedef __attribute__((ext_vector_type(8)))  float    v8f;

// k index inside a 16x32 (MxK) f16 A/B fragment:
//   lanes 0-15  (hi=0): element i -> K = {0..7, 16..23}
//   lanes 16-31 (hi=1): element i -> K = {8..15, 24..31}
__device__ __forceinline__ int frag_k(int kbase, int hi, int i) {
    return kbase + hi * 8 + (i < 8 ? i : i + 8);
}

// Load one B fragment (KxN weight, row-major [K x N] f32 in global memory)
__device__ __forceinline__ v16h load_b_frag(const float* __restrict__ W, int N,
                                            int n, int kbase, int hi) {
    v16h b;
#pragma unroll
    for (int i = 0; i < 16; ++i)
        b[i] = (_Float16)W[frag_k(kbase, hi, i) * N + n];
    return b;
}

// ---------------------------------------------------------------------------
// Kernel 1: edge messages + scatter (sum / max / count)
//   msg[e,:] = x[src[e],:] @ W + bias     (16 edges per wave-tile, WMMA)
//   sum[dst[e],:]   += msg
//   max[dst[e],:]    = max(., msg)
//     Max trick: accumulator init = 0 (int bits of +0.0). For v >= 0 the
//     float ordering equals the signed-int ordering of its bits; for v < 0
//     the bits form a negative int which can never raise a >=0 accumulator.
//     So an unconditional signed atomicMax on the raw bits computes exactly
//     max(0, segment_max) -- branch-free.
//   cnt[dst[e]]     += 1
// ---------------------------------------------------------------------------
__global__ void __launch_bounds__(256, 1)
edge_msg_scatter(const float* __restrict__ x,
                 const float* __restrict__ W,     // [64,64]
                 const float* __restrict__ bias,  // [64]
                 const int*   __restrict__ src,
                 const int*   __restrict__ dst,
                 int          E,
                 float* __restrict__ sum_out,
                 int*   __restrict__ max_out,
                 float* __restrict__ cnt_out) {
    const int lane = threadIdx.x & 31;
    const int hi   = lane >> 4;
    const int nl   = lane & 15;
    const int wave   = (blockIdx.x * blockDim.x + threadIdx.x) >> 5;
    const int nwaves = (gridDim.x * blockDim.x) >> 5;

    // Preload W as 4 (N) x 2 (K) fragments into registers: reused every tile.
    v16h Bf[4][2];
#pragma unroll
    for (int nt = 0; nt < 4; ++nt)
#pragma unroll
        for (int kt = 0; kt < 2; ++kt)
            Bf[nt][kt] = load_b_frag(W, DIM, nt * 16 + nl, kt * 32, hi);
    float bv[4];
#pragma unroll
    for (int nt = 0; nt < 4; ++nt) bv[nt] = bias[nt * 16 + nl];

    const int ntiles = E >> 4;  // E divisible by 16
    for (int t = wave; t < ntiles; t += nwaves) {
        const int erow = t * 16 + nl;  // A-layout row owned by this lane pair
        const int s = src[erow];
        const float* row = x + (long)s * DIM;

        v16h A[2];
#pragma unroll
        for (int kt = 0; kt < 2; ++kt)
#pragma unroll
            for (int i = 0; i < 16; ++i)
                A[kt][i] = (_Float16)row[frag_k(kt * 32, hi, i)];

        int dbase[8];  // element offset of row start in the accumulators
#pragma unroll
        for (int r = 0; r < 8; ++r) dbase[r] = dst[t * 16 + hi * 8 + r] * DIM;

#pragma unroll
        for (int nt = 0; nt < 4; ++nt) {
            v8f c = {};
            c = __builtin_amdgcn_wmma_f32_16x16x32_f16(false, A[0], false, Bf[nt][0],
                                                       (short)0, c, false, false);
            c = __builtin_amdgcn_wmma_f32_16x16x32_f16(false, A[1], false, Bf[nt][1],
                                                       (short)0, c, false, false);
            const int col = nt * 16 + nl;
#pragma unroll
            for (int r = 0; r < 8; ++r) {
                float v = c[r] + bv[nt];
                int off = dbase[r] + col;
                atomicAdd(&sum_out[off], v);
                atomicMax(&max_out[off], __float_as_int(v));  // see note above
            }
        }
        if (hi == 0) atomicAdd(&cnt_out[dst[erow]], 1.0f);
    }
}

// total[node, k] for k in [0,192): [mean(64) | sum(64) | max(64)]
__device__ __forceinline__ float tot_val(const float* __restrict__ sum_in,
                                         const int*   __restrict__ max_in,
                                         int node, int k, float cinv) {
    if (k < 64)  return sum_in[node * DIM + k] * cinv;
    if (k < 128) return sum_in[node * DIM + (k - 64)];
    return __int_as_float(max_in[node * DIM + (k - 128)]);  // non-negative
}

// ---------------------------------------------------------------------------
// Kernel 2: node update (attention gate, aggregation, selfloop)
// One wave processes a 16-node tile with three WMMA GEMMs.
// ---------------------------------------------------------------------------
#define WPB 4  // waves per block

__global__ void __launch_bounds__(32 * WPB, 1)
node_update(const float* __restrict__ sum_in,
            const int*   __restrict__ max_in,
            const float* __restrict__ cnt_in,
            const float* __restrict__ x,       // [N,64]
            const float* __restrict__ W_att,   // [192,192]
            const float* __restrict__ b_att,   // [192]
            const float* __restrict__ W_aggr,  // [192,64]
            const float* __restrict__ b_aggr,  // [64]
            const float* __restrict__ W_self,  // [128,64]
            const float* __restrict__ b_self,  // [64]
            float* __restrict__ out,           // [N,64]
            int Nn) {
    const int lane = threadIdx.x & 31;
    const int hi   = lane >> 4;
    const int nl   = lane & 15;
    const int wib  = threadIdx.x >> 5;
    const int wave   = (blockIdx.x * blockDim.x + threadIdx.x) >> 5;
    const int nwaves = (gridDim.x * blockDim.x) >> 5;

    __shared__ float lds[WPB][16 * 192];
    float* g = lds[wib];

    const int ntile_count = Nn >> 4;  // both N divisible by 16
    for (int t = wave; t < ntile_count; t += nwaves) {
        const int anode = t * 16 + nl;  // A-layout row node
        const float acinv = 1.0f / fmaxf(cnt_in[anode], 1.0f);

        // A fragments of `total` (K = 192 -> 6 k-tiles)
        v16h At[6];
#pragma unroll
        for (int kt = 0; kt < 6; ++kt)
#pragma unroll
            for (int i = 0; i < 16; ++i)
                At[kt][i] = (_Float16)tot_val(sum_in, max_in, anode,
                                              frag_k(kt * 32, hi, i), acinv);

        // C-layout row metadata
        int   cnode[8];
        float ccinv[8];
        bool  cmask[8];
#pragma unroll
        for (int r = 0; r < 8; ++r) {
            int node = t * 16 + hi * 8 + r;
            cnode[r] = node;
            float cc = cnt_in[node];
            ccinv[r] = 1.0f / fmaxf(cc, 1.0f);
            cmask[r] = (cc == 0.0f);  // no incoming messages
        }

        // GEMM1: att = sigmoid(total @ W_att + b_att); g = total * att -> LDS
#pragma unroll
        for (int nt = 0; nt < 12; ++nt) {
            v8f c = {};
#pragma unroll
            for (int kt = 0; kt < 6; ++kt) {
                v16h b = load_b_frag(W_att, 192, nt * 16 + nl, kt * 32, hi);
                c = __builtin_amdgcn_wmma_f32_16x16x32_f16(false, At[kt], false, b,
                                                           (short)0, c, false, false);
            }
            const int col = nt * 16 + nl;
            const float bb = b_att[col];
#pragma unroll
            for (int r = 0; r < 8; ++r) {
                float att = 1.0f / (1.0f + __expf(-(c[r] + bb)));
                float tv  = tot_val(sum_in, max_in, cnode[r], col, ccinv[r]);
                g[(hi * 8 + r) * 192 + col] = tv * att;
            }
        }

        // GEMM2: o = g @ W_aggr  (192 -> 64); g consumed via register
        // fragments first, so each n-tile can be retired to LDS immediately.
        v16h Ag[6];
#pragma unroll
        for (int kt = 0; kt < 6; ++kt)
#pragma unroll
            for (int i = 0; i < 16; ++i)
                Ag[kt][i] = (_Float16)g[nl * 192 + frag_k(kt * 32, hi, i)];

#pragma unroll
        for (int nt = 0; nt < 4; ++nt) {
            v8f c = {};
#pragma unroll
            for (int kt = 0; kt < 6; ++kt) {
                v16h b = load_b_frag(W_aggr, DIM, nt * 16 + nl, kt * 32, hi);
                c = __builtin_amdgcn_wmma_f32_16x16x32_f16(false, Ag[kt], false, b,
                                                           (short)0, c, false, false);
            }
            // bias + mask + leaky_relu, stage conv output (16x64) into LDS
            const int col = nt * 16 + nl;
            const float bb = b_aggr[col];
#pragma unroll
            for (int r = 0; r < 8; ++r) {
                float v = cmask[r] ? 0.0f : (c[r] + bb);
                v = v > 0.0f ? v : 0.01f * v;  // leaky_relu
                g[(hi * 8 + r) * 64 + col] = v;
            }
        }

        // GEMM3: y = leaky_relu([out | x] @ W_self + b_self)  (K = 128)
        v16h As[4];
#pragma unroll
        for (int kt = 0; kt < 4; ++kt)
#pragma unroll
            for (int i = 0; i < 16; ++i) {
                int k = frag_k(kt * 32, hi, i);
                float v = (k < 64) ? g[nl * 64 + k] : x[anode * DIM + (k - 64)];
                As[kt][i] = (_Float16)v;
            }

#pragma unroll
        for (int nt = 0; nt < 4; ++nt) {
            v8f c = {};
#pragma unroll
            for (int kt = 0; kt < 4; ++kt) {
                v16h b = load_b_frag(W_self, DIM, nt * 16 + nl, kt * 32, hi);
                c = __builtin_amdgcn_wmma_f32_16x16x32_f16(false, As[kt], false, b,
                                                           (short)0, c, false, false);
            }
            const int col = nt * 16 + nl;
            const float bb = b_self[col];
#pragma unroll
            for (int r = 0; r < 8; ++r) {
                float y = c[r] + bb;
                y = y > 0.0f ? y : 0.01f * y;
                float xv = x[cnode[r] * DIM + col];
                out[cnode[r] * DIM + col] = cmask[r] ? xv : y;  // masked fallback
            }
        }
    }
}

// ---------------------------------------------------------------------------
// Host launcher
// ---------------------------------------------------------------------------
extern "C" void kernel_launch(void* const* d_in, const int* in_sizes, int n_in,
                              void* d_out, int out_size, void* d_ws, size_t ws_size,
                              hipStream_t stream) {
    const float* x_track     = (const float*)d_in[0];
    const float* x_inter     = (const float*)d_in[1];
    const float* W_conv_ti   = (const float*)d_in[2];
    const float* b_conv_ti   = (const float*)d_in[3];
    const float* W_conv_it   = (const float*)d_in[4];
    const float* b_conv_it   = (const float*)d_in[5];
    const float* W_att_track = (const float*)d_in[6];
    const float* b_att_track = (const float*)d_in[7];
    const float* W_aggr_track= (const float*)d_in[8];
    const float* b_aggr_track= (const float*)d_in[9];
    const float* W_att_inter = (const float*)d_in[10];
    const float* b_att_inter = (const float*)d_in[11];
    const float* W_aggr_inter= (const float*)d_in[12];
    const float* b_aggr_inter= (const float*)d_in[13];
    const float* W_self_track= (const float*)d_in[14];
    const float* b_self_track= (const float*)d_in[15];
    const float* W_self_inter= (const float*)d_in[16];
    const float* b_self_inter= (const float*)d_in[17];
    const int*   src_ti      = (const int*)d_in[18];
    const int*   dst_ti      = (const int*)d_in[19];
    const int*   src_it      = (const int*)d_in[20];
    const int*   dst_it      = (const int*)d_in[21];

    float* outp = (float*)d_out;

    // Workspace layout (f32 elements)
    float* sum_i = (float*)d_ws;                     // N_INTER*64
    int*   mx_i  = (int*)(sum_i + (size_t)N_INTER * DIM);
    float* cnt_i = (float*)(mx_i + (size_t)N_INTER * DIM);
    float* sum_t = cnt_i + N_INTER;                  // N_TRACK*64
    int*   mx_t  = (int*)(sum_t + (size_t)N_TRACK * DIM);
    float* cnt_t = (float*)(mx_t + (size_t)N_TRACK * DIM);
    const size_t ws_elems = (size_t)(N_INTER + N_TRACK) * (2 * DIM + 1);

    // Zero accumulators every call (capture-safe memset node)
    hipMemsetAsync(d_ws, 0, ws_elems * sizeof(float), stream);

    // Edge phase: track -> inter, inter -> track
    edge_msg_scatter<<<1024, 256, 0, stream>>>(x_track, W_conv_ti, b_conv_ti,
                                               src_ti, dst_ti, N_EDGE,
                                               sum_i, mx_i, cnt_i);
    edge_msg_scatter<<<1024, 256, 0, stream>>>(x_inter, W_conv_it, b_conv_it,
                                               src_it, dst_it, N_EDGE,
                                               sum_t, mx_t, cnt_t);

    // Node phase: track nodes -> d_out[0:N_TRACK*64], inter -> after
    node_update<<<512, 32 * WPB, 0, stream>>>(sum_t, mx_t, cnt_t, x_track,
                                              W_att_track, b_att_track,
                                              W_aggr_track, b_aggr_track,
                                              W_self_track, b_self_track,
                                              outp, N_TRACK);
    node_update<<<512, 32 * WPB, 0, stream>>>(sum_i, mx_i, cnt_i, x_inter,
                                              W_att_inter, b_att_inter,
                                              W_aggr_inter, b_aggr_inter,
                                              W_self_inter, b_self_inter,
                                              outp + (size_t)N_TRACK * DIM, N_INTER);
}
